// HybridDecoderBlock_43370579755492
// MI455X (gfx1250) — matrix operand order repaired and checked by reading, not compiled, not run on hardware
//
#include <hip/hip_runtime.h>
#include <hip/hip_bf16.h>
#include <math.h>
#include <stdint.h>

typedef __attribute__((ext_vector_type(16))) _Float16 v16h;
typedef __attribute__((ext_vector_type(8)))  float    v8f;

#define HW 4096          // 64*64
#define NROWS 131072     // 4096 * 32 sequence rows
#define CH 64
#define DI 128

// ---------------------------------------------------------------------------
// Repack conv weights [co][ci][tap(18)] -> [co][tap][ci] so B fragments are
// contiguous 8-float runs (b128 clauses).
// ---------------------------------------------------------------------------
__global__ void repack_w_kernel(const float* __restrict__ wgt, float* __restrict__ wr, int Cin) {
    int Kred = Cin * 18;
    int idx = blockIdx.x * 256 + threadIdx.x;
    if (idx >= 64 * Kred) return;
    int co = idx / Kred; int k = idx - co * Kred;   // repacked k = tap*Cin + ci
    int tap = k / Cin;   int ci = k - tap * Cin;
    wr[idx] = wgt[(size_t)co * Kred + ci * 18 + tap];
}

// ---------------------------------------------------------------------------
// Implicit-GEMM 3D conv, tap-major, compile-time Cin/Zin so A-fragment offsets
// fold into global_load immediate offsets (clause-friendly, no address VALU).
// zi/yi tap skips are wave-uniform (EXEC all-ones at WMMA); x-border handled
// branchlessly via clamp + 0/1 multiplier.
// ---------------------------------------------------------------------------
template<int Cin, int Zin>
__global__ void conv3d_igemm_kernel(const float* __restrict__ in, const float* __restrict__ wr,
                                    float* __restrict__ out, int Zout) {
    constexpr size_t ZHW  = (size_t)Zin * 4096;   // ci stride (elements)
    constexpr int    Kred = Cin * 18;

    int wid  = blockIdx.x * (blockDim.x >> 5) + (threadIdx.x >> 5);
    int lane = threadIdx.x & 31;
    int l15  = lane & 15, hi = lane >> 4;
    int cot = wid & 3;  wid >>= 2;
    int xt  = wid & 3;  wid >>= 2;
    int y   = wid & 63; wid >>= 6;
    int zo  = wid;
    if (zo >= Zout) return;                 // wave-uniform
    int x = xt * 16 + l15;

    // Fold the per-lane half-wave ci offset (+8*hi) into the A base pointer.
    const float* in_h = in + (size_t)(8 * hi) * ZHW;
    const float* brow = wr + (size_t)(cot*16 + l15) * Kred + 8*hi;

    v8f acc = {};
    for (int kz = 0; kz < 2; ++kz) {
        int zi = zo - 1 + kz;
        if ((unsigned)zi >= (unsigned)Zin) continue;     // uniform
        for (int ky = 0; ky < 3; ++ky) {
            int yi = y - 1 + ky;
            if ((unsigned)yi >= 64u) continue;           // uniform
            #pragma unroll
            for (int kx = 0; kx < 3; ++kx) {
                int xi = x - 1 + kx;
                float sel = ((unsigned)xi < 64u) ? 1.f : 0.f;
                int xic = xi < 0 ? 0 : (xi > 63 ? 63 : xi);
                const float* abase = in_h + (((size_t)zi) << 12) + (yi << 6) + xic;
                int tap = (kz*3 + ky)*3 + kx;
                const float* bbase = brow + tap * Cin;
                #pragma unroll
                for (int cb = 0; cb < Cin; cb += 32) {
                    const float* ac = abase + (size_t)cb * ZHW;
                    const float* wp = bbase + cb;
                    v16h a, b;
                    #pragma unroll
                    for (int r = 0; r < 8; ++r) {
                        // compile-time offsets: {0..7, 16..23} * ZHW
                        constexpr int KO0[8] = {0, 2, 4, 6, 16, 18, 20, 22};
                        size_t o = (size_t)KO0[r] * ZHW;
                        a[2*r]   = (_Float16)(ac[o]       * sel);
                        a[2*r+1] = (_Float16)(ac[o + ZHW] * sel);
                    }
                    #pragma unroll
                    for (int j = 0; j < 16; ++j) b[j] = (_Float16)wp[(j & 7) + 16 * (j >> 3)];
                    acc = __builtin_amdgcn_wmma_f32_16x16x32_f16(false, a, false, b,
                                                                 (short)0, acc, false, false);
                }
            }
        }
    }
    int co = cot * 16 + l15;
    #pragma unroll
    for (int r = 0; r < 8; ++r) {
        int xx = xt * 16 + r + 8*hi;
        out[(((size_t)co * Zout + zo) << 12) + (y << 6) + xx] = acc[r];
    }
}

// ---------------------------------------------------------------------------
// Per-channel mean / rstd reduction (InstanceNorm stats). One block per channel.
// ---------------------------------------------------------------------------
__global__ void stats_kernel(const float* __restrict__ x, size_t chanStride, size_t itemStride,
                             int count, float* __restrict__ stats) {
    int c = blockIdx.x;
    const float* p = x + (size_t)c * chanStride;
    float s = 0.f, q = 0.f;
    for (int i = threadIdx.x; i < count; i += blockDim.x) {
        float v = p[(size_t)i * itemStride];
        s += v; q += v * v;
    }
    __shared__ float ss[256], sq[256];
    ss[threadIdx.x] = s; sq[threadIdx.x] = q;
    __syncthreads();
    for (int w = 128; w > 0; w >>= 1) {
        if (threadIdx.x < (unsigned)w) { ss[threadIdx.x] += ss[threadIdx.x + w]; sq[threadIdx.x] += sq[threadIdx.x + w]; }
        __syncthreads();
    }
    if (threadIdx.x == 0) {
        float m = ss[0] / count;
        float v = sq[0] / count - m * m;
        stats[2*c]   = m;
        stats[2*c+1] = rsqrtf(v + 1e-5f);
    }
}

// InstanceNorm apply + LeakyReLU(0.01), with optional depth slice Zs -> Zd
__global__ void norm_leaky_kernel(const float* __restrict__ src, float* __restrict__ dst,
                                  const float* __restrict__ stats, const float* __restrict__ g,
                                  const float* __restrict__ b, int Zs, int Zd) {
    int idx = blockIdx.x * 256 + threadIdx.x;
    int total = CH * Zd * HW;
    if (idx >= total) return;
    int c = idx / (Zd * HW); int rem = idx - c * (Zd * HW);
    int z = rem >> 12; int p = rem & 4095;
    float v  = src[((size_t)c * Zs + z) * HW + p];
    float yn = (v - stats[2*c]) * stats[2*c+1] * g[c] + b[c];
    dst[((size_t)c * Zd + z) * HW + p] = yn > 0.f ? yn : 0.01f * yn;
}

// Depthwise 1x3x3 conv, pad (0,1,1)
__global__ void dwconv_kernel(const float* __restrict__ h, const float* __restrict__ w,
                              float* __restrict__ out) {
    int idx = blockIdx.x * 256 + threadIdx.x;
    if (idx >= CH * 32 * HW) return;
    int p = idx & 4095; int z = (idx >> 12) & 31; int c = idx >> 17;
    int y = p >> 6, x = p & 63;
    const float* wc = w + c * 9;
    const float* hp = h + ((size_t)c * 32 + z) * HW;
    float s = 0.f;
    #pragma unroll
    for (int ky = 0; ky < 3; ++ky) {
        int yi = y - 1 + ky;
        if ((unsigned)yi >= 64u) continue;
        #pragma unroll
        for (int kx = 0; kx < 3; ++kx) {
            int xi = x - 1 + kx;
            if ((unsigned)xi >= 64u) continue;
            s += wc[ky*3 + kx] * hp[(yi << 6) + xi];
        }
    }
    out[idx] = s;
}

// Pointwise (1x1x1) conv as WMMA GEMM over channel-major layout (64 x 131072)
__global__ void pw_gemm_kernel(const float* __restrict__ in, const float* __restrict__ w,
                               float* __restrict__ out) {
    int wid  = blockIdx.x * (blockDim.x >> 5) + (threadIdx.x >> 5);
    int lane = threadIdx.x & 31;
    int l15  = lane & 15, hi = lane >> 4;
    int tn = wid & 3; int tm = wid >> 2;
    if (tm * 16 >= NROWS) return;
    int p = tm * 16 + l15;
    const float* in_h = in + (size_t)(8 * hi) * NROWS + p;
    v8f acc = {};
    #pragma unroll
    for (int kb = 0; kb < 64; kb += 32) {
        v16h a, b;
        const float* ac = in_h + (size_t)kb * NROWS;
        #pragma unroll
        for (int r = 0; r < 8; ++r) {
            constexpr int KO0[8] = {0, 2, 4, 6, 16, 18, 20, 22};
            size_t o = (size_t)KO0[r] * NROWS;
            a[2*r]   = (_Float16)ac[o];
            a[2*r+1] = (_Float16)ac[o + NROWS];
        }
        int n = tn * 16 + l15;
        const float* wrow = w + (size_t)n * 64 + kb + 8*hi;
        #pragma unroll
        for (int r = 0; r < 8; ++r) { int kk = r < 4 ? 2*r : 2*r + 8; b[2*r] = (_Float16)wrow[kk]; b[2*r+1] = (_Float16)wrow[kk + 1]; }
        acc = __builtin_amdgcn_wmma_f32_16x16x32_f16(false, a, false, b, (short)0, acc, false, false);
    }
    int co = tn * 16 + l15;
    #pragma unroll
    for (int r = 0; r < 8; ++r)
        out[(size_t)co * NROWS + tm * 16 + r + 8*hi] = acc[r];
}

// x2 = h + SiLU(InstanceNorm(pw_out))
__global__ void combine_kernel(const float* __restrict__ h, const float* __restrict__ pw,
                               const float* __restrict__ stats, const float* __restrict__ g,
                               const float* __restrict__ b, float* __restrict__ x2) {
    int idx = blockIdx.x * 256 + threadIdx.x;
    if (idx >= CH * 32 * HW) return;
    int c = idx >> 17;
    float v  = pw[idx];
    float yn = (v - stats[2*c]) * stats[2*c+1] * g[c] + b[c];
    x2[idx] = h[idx] + yn / (1.f + __expf(-yn));
}

// Build LayerNorm'd sequence rows: seq[row, c], row = (hh*64+ww)*32 + z
__global__ void seq_ln_kernel(const float* __restrict__ x2, const float* __restrict__ g,
                              const float* __restrict__ b, float* __restrict__ seq) {
    int row = blockIdx.x * 256 + threadIdx.x;
    if (row >= NROWS) return;
    int z = row & 31; int pair = row >> 5;
    const float* base = x2 + ((size_t)z << 12) + pair;   // + c * 131072
    float s = 0.f, q = 0.f;
    #pragma unroll 4
    for (int c = 0; c < 64; ++c) { float v = base[(size_t)c << 17]; s += v; q += v * v; }
    float m = s * (1.f/64.f);
    float r = rsqrtf(q * (1.f/64.f) - m*m + 1e-5f);
    float* o = seq + (size_t)row * 64;
    #pragma unroll 4
    for (int c = 0; c < 64; ++c) {
        float v = base[(size_t)c << 17];
        o[c] = (v - m) * r * g[c] + b[c];
    }
}

// Generic WMMA GEMM: out[M x N] = A[M x K] @ W[N x K]^T (+bias). M % 16 == 0; N masked.
__global__ void gemm_wmma_kernel(const float* __restrict__ A, const float* __restrict__ W,
                                 const float* __restrict__ bias, float* __restrict__ out,
                                 int M, int N, int K, int ntiles) {
    int wid  = blockIdx.x * (blockDim.x >> 5) + (threadIdx.x >> 5);
    int lane = threadIdx.x & 31;
    int l15  = lane & 15, hi = lane >> 4;
    int tn = wid % ntiles;
    int tm = wid / ntiles;
    if (tm * 16 >= M) return;                 // wave-uniform
    int mbase = tm * 16, nbase = tn * 16;
    int n = nbase + l15;
    const float* arow = A + (size_t)(mbase + l15) * K;
    const float* wrow = (n < N) ? (W + (size_t)n * K + 8*hi) : nullptr;

    v8f acc = {};
    for (int kb = 0; kb < K; kb += 32) {
        v16h a, b;
        __builtin_prefetch(arow + kb + 64, 0, 1);
        #pragma unroll
        for (int r = 0; r < 8; ++r) {
            int ko = (r < 4 ? 2*r : 16 + 2*(r-4)) + 8*hi;
            a[2*r]   = (_Float16)arow[kb + ko];
            a[2*r+1] = (_Float16)arow[kb + ko + 1];
        }
        if (n < N) {
            #pragma unroll
            for (int r = 0; r < 8; ++r) { int kk = r < 4 ? 2*r : 2*r + 8; b[2*r] = (_Float16)wrow[kb + kk]; b[2*r+1] = (_Float16)wrow[kb + kk + 1]; }
        } else {
            #pragma unroll
            for (int r = 0; r < 16; ++r) b[r] = (_Float16)0.f;
        }
        acc = __builtin_amdgcn_wmma_f32_16x16x32_f16(false, a, false, b, (short)0, acc, false, false);
    }
    if (n < N) {
        #pragma unroll
        for (int r = 0; r < 8; ++r) {
            int m = mbase + r + 8*hi;
            float v = acc[r];
            if (bias) v += bias[n];
            out[(size_t)m * N + n] = v;
        }
    }
}

// Causal depthwise conv1d (kernel 4, left-pad 3) + bias + SiLU.  x_in = xz[..., :128]
__global__ void conv1d_silu_kernel(const float* __restrict__ xz, const float* __restrict__ w,
                                   const float* __restrict__ bias, float* __restrict__ xa) {
    int idx = blockIdx.x * 256 + threadIdx.x;
    if (idx >= 4096 * DI) return;
    int c = idx & 127; int nseq = idx >> 7;
    const float* wc = w + c * 4;
    float w0 = wc[0], w1 = wc[1], w2 = wc[2], w3 = wc[3], bb = bias[c];
    float x0 = 0.f, x1 = 0.f, x2 = 0.f;
    for (int t = 0; t < 32; ++t) {
        size_t row = (size_t)nseq * 32 + t;
        float x3 = xz[row * 256 + c];
        float s  = w0*x0 + w1*x1 + w2*x2 + w3*x3 + bb;
        xa[row * 128 + c] = s / (1.f + __expf(-s));
        x0 = x1; x1 = x2; x2 = x3;
    }
}

// dt = softplus(x_dbl[:, :4] @ dt_proj_w^T + dt_proj_b)   (K=4, VALU kernel)
__global__ void dt_kernel(const float* __restrict__ xdbl, const float* __restrict__ w,
                          const float* __restrict__ b, float* __restrict__ dt) {
    size_t idx = (size_t)blockIdx.x * 256 + threadIdx.x;
    if (idx >= (size_t)NROWS * DI) return;
    int d = (int)(idx & 127); size_t row = idx >> 7;
    const float* xr = xdbl + row * 36;
    const float* wr = w + d * 4;
    float v = b[d] + xr[0]*wr[0] + xr[1]*wr[1] + xr[2]*wr[2] + xr[3]*wr[3];
    dt[idx] = v > 20.f ? v : log1pf(__expf(v));
}

// Selective scan over T=32, D_STATE=16; fused +D*u and SiLU(z) gating.
// One block per sequence (128 threads = one per channel d). dt/y may alias.
// Wave 0 broadcasts B_t/C_t into LDS via CDNA5 async global->LDS copy (ASYNCcnt).
__global__ void scan_kernel(const float* __restrict__ xdbl, const float* __restrict__ xa,
                            const float* __restrict__ zbuf, const float* __restrict__ dt,
                            const float* __restrict__ A_log, const float* __restrict__ Dp,
                            float* __restrict__ y) {
    int nseq = blockIdx.x, d = threadIdx.x;
    float Arow[16], st[16];
    #pragma unroll
    for (int s = 0; s < 16; ++s) { Arow[s] = -__expf(A_log[d*16 + s]); st[s] = 0.f; }
    float dp = Dp[d];
    __shared__ float bc[32];
    for (int t = 0; t < 32; ++t) {
        size_t row = (size_t)nseq * 32 + t;
        if (d < 32) {   // wave-uniform: only wave 0 enters
            uint32_t ldsaddr = (uint32_t)(uintptr_t)(&bc[d]);
            uint64_t gaddr   = (uint64_t)(uintptr_t)(xdbl + row * 36 + 4 + d); // B_t, C_t
            asm volatile("global_load_async_to_lds_b32 %0, %1, off"
                         :: "v"(ldsaddr), "v"(gaddr) : "memory");
            asm volatile("s_wait_asynccnt 0x0" ::: "memory");
        }
        __syncthreads();
        float dtv = dt[row * 128 + d];
        float u   = xa[row * 128 + d];
        float du  = dtv * u;
        float acc = 0.f;
        #pragma unroll
        for (int s = 0; s < 16; ++s) {
            st[s] = __expf(dtv * Arow[s]) * st[s] + du * bc[s];
            acc  += st[s] * bc[16 + s];
        }
        float zv = zbuf[row * 256 + 128 + d];
        float yv = (acc + dp * u) * (zv / (1.f + __expf(-zv)));
        y[row * 128 + d] = yv;
        __syncthreads();
    }
}

// Final: out[c,z,p] = residual[c,z,p] + InstanceNorm(o2)[c,z,p]
__global__ void final_kernel(const float* __restrict__ res, const float* __restrict__ o2,
                             const float* __restrict__ stats, const float* __restrict__ g,
                             const float* __restrict__ b, float* __restrict__ out) {
    int idx = blockIdx.x * 256 + threadIdx.x;
    if (idx >= CH * 32 * HW) return;
    int p = idx & 4095; int z = (idx >> 12) & 31; int c = idx >> 17;
    size_t row = ((size_t)p << 5) + z;
    float v = o2[row * 64 + c];
    out[idx] = res[idx] + (v - stats[2*c]) * stats[2*c+1] * g[c] + b[c];
}

// ---------------------------------------------------------------------------
extern "C" void kernel_launch(void* const* d_in, const int* in_sizes, int n_in,
                              void* d_out, int out_size, void* d_ws, size_t ws_size,
                              hipStream_t stream) {
    const float* x        = (const float*)d_in[0];
    const float* c1_w     = (const float*)d_in[1];
    const float* in1_g    = (const float*)d_in[2];
    const float* in1_b    = (const float*)d_in[3];
    const float* c2_w     = (const float*)d_in[4];
    const float* in2_g    = (const float*)d_in[5];
    const float* in2_b    = (const float*)d_in[6];
    const float* dw_w     = (const float*)d_in[7];
    const float* pw_w     = (const float*)d_in[8];
    const float* inr_g    = (const float*)d_in[9];
    const float* inr_b    = (const float*)d_in[10];
    const float* ln_g     = (const float*)d_in[11];
    const float* ln_b     = (const float*)d_in[12];
    const float* in_proj  = (const float*)d_in[13];
    const float* c1d_w    = (const float*)d_in[14];
    const float* c1d_b    = (const float*)d_in[15];
    const float* xproj_w  = (const float*)d_in[16];
    const float* dtp_w    = (const float*)d_in[17];
    const float* dtp_b    = (const float*)d_in[18];
    const float* A_log    = (const float*)d_in[19];
    const float* D_p      = (const float*)d_in[20];
    const float* outp_w   = (const float*)d_in[21];
    const float* proj_w   = (const float*)d_in[22];
    const float* proj_b   = (const float*)d_in[23];
    const float* pn_g     = (const float*)d_in[24];
    const float* pn_b     = (const float*)d_in[25];

    char* w = (char*)d_ws;
    float* A  = (float*)(w + 0);              // h1raw (34.6MB) -> dw out -> x2 -> o2
    float* Bb = (float*)(w + 35651584ull);    // h2raw (34MB) -> pw out -> seq -> o1
    float* Cb = (float*)(w + 71303168ull);    // residual h (32MB)
    float* Dx = (float*)(w + 104857600ull);   // xz  (128MB)
    float* E  = (float*)(w + 239075328ull);   // xa  (64MB)
    float* F  = (float*)(w + 306184192ull);   // x_dbl (18MB)
    float* G  = (float*)(w + 325058560ull);   // dt -> y (64MB)
    float* ST = (float*)(w + 392167424ull);   // stats (2*64 floats)
    float* WR = (float*)(w + 392167936ull);   // repacked conv weights (<= 288KB)

    // conv1 (IC=32, K=576) -> A, InstanceNorm+leaky in place
    repack_w_kernel<<<dim3(144), dim3(256), 0, stream>>>(c1_w, WR, 32);
    conv3d_igemm_kernel<32, 32><<<dim3(4224), dim3(256), 0, stream>>>(x, WR, A, 33);
    stats_kernel<<<dim3(64), dim3(256), 0, stream>>>(A, (size_t)33*HW, 1, 33*HW, ST);
    norm_leaky_kernel<<<dim3(33792), dim3(256), 0, stream>>>(A, A, ST, in1_g, in1_b, 33, 33);

    // conv2 (IC=64, K=1152) -> Bb, InstanceNorm(full 34)+leaky, slice to Z=32 -> Cb (residual h)
    repack_w_kernel<<<dim3(288), dim3(256), 0, stream>>>(c2_w, WR, 64);
    conv3d_igemm_kernel<64, 33><<<dim3(4352), dim3(256), 0, stream>>>(A, WR, Bb, 34);
    stats_kernel<<<dim3(64), dim3(256), 0, stream>>>(Bb, (size_t)34*HW, 1, 34*HW, ST);
    norm_leaky_kernel<<<dim3(32768), dim3(256), 0, stream>>>(Bb, Cb, ST, in2_g, in2_b, 34, 32);

    // depthwise -> A; pointwise (WMMA) -> Bb; x2 = h + SiLU(IN(pw)) -> A
    dwconv_kernel<<<dim3(32768), dim3(256), 0, stream>>>(Cb, dw_w, A);
    pw_gemm_kernel<<<dim3(8192), dim3(128), 0, stream>>>(A, pw_w, Bb);
    stats_kernel<<<dim3(64), dim3(256), 0, stream>>>(Bb, (size_t)NROWS, 1, NROWS, ST);
    combine_kernel<<<dim3(32768), dim3(256), 0, stream>>>(Cb, Bb, ST, inr_g, inr_b, A);

    // LayerNorm'd sequence rows -> Bb; in_proj GEMM (N=256, K=64) -> Dx (xz)
    seq_ln_kernel<<<dim3(512), dim3(256), 0, stream>>>(A, ln_g, ln_b, Bb);
    gemm_wmma_kernel<<<dim3(32768), dim3(128), 0, stream>>>(Bb, in_proj, nullptr, Dx,
                                                            NROWS, 256, 64, 16);

    // causal conv1d + SiLU -> E (xa); x_proj GEMM (N=36, K=128) -> F
    conv1d_silu_kernel<<<dim3(2048), dim3(256), 0, stream>>>(Dx, c1d_w, c1d_b, E);
    gemm_wmma_kernel<<<dim3(6144), dim3(128), 0, stream>>>(E, xproj_w, nullptr, F,
                                                           NROWS, 36, 128, 3);

    // dt = softplus(...) -> G; selective scan (reads dt, writes y in place over G)
    dt_kernel<<<dim3(65536), dim3(256), 0, stream>>>(F, dtp_w, dtp_b, G);
    scan_kernel<<<dim3(4096), dim3(128), 0, stream>>>(F, E, Dx, G, A_log, D_p, G);

    // out_proj (N=64, K=128) -> Bb; proj (+bias, N=64, K=64) -> A
    gemm_wmma_kernel<<<dim3(8192), dim3(128), 0, stream>>>(G, outp_w, nullptr, Bb,
                                                           NROWS, 64, 128, 4);
    gemm_wmma_kernel<<<dim3(8192), dim3(128), 0, stream>>>(Bb, proj_w, proj_b, A,
                                                           NROWS, 64, 64, 4);

    // final InstanceNorm over columns of A + residual -> d_out
    stats_kernel<<<dim3(64), dim3(256), 0, stream>>>(A, 1, 64, NROWS, ST);
    final_kernel<<<dim3(32768), dim3(256), 0, stream>>>(Cb, A, ST, pn_g, pn_b, (float*)d_out);
}
